// MultiheadGCNLayer_45019847197092
// MI455X (gfx1250) — matrix operand order repaired and checked by reading, not compiled, not run on hardware
//
#include <hip/hip_runtime.h>
#include <hip/hip_bf16.h>

// MultiheadGCNLayer fused: W-collapse -> GEMM1 (f16 WMMA) -> GEMM2 (f16 WMMA)
// with fused adjacency row-sum normalization.
// CDNA5 paths: v_wmma_f32_16x16x32_f16, global_load_async_to_lds_b128 with
// shared-address offset immediates, ASYNCcnt-depth-2 software pipeline.

typedef __attribute__((ext_vector_type(16))) _Float16 v16h;
typedef __attribute__((ext_vector_type(8)))  _Float16 v8h;
typedef __attribute__((ext_vector_type(8)))  float    v8f;
typedef __attribute__((ext_vector_type(4)))  float    f4;

#define B_    8
#define N_    2048
#define CIN   256
#define COUT  256
#define F_    8
#define EPS_  1e-4f

static __device__ __forceinline__ v16h cat8(v8h lo, v8h hi) {
  return __builtin_shufflevector(lo, hi, 0,1,2,3,4,5,6,7,8,9,10,11,12,13,14,15);
}

// A fragment (16x32 f16): LDS tile row-major [row][32 halves].
// Lane l holds row M=l&15; halves 0..7 = K[ks..ks+7], halves 8..15 = K[16+ks..],
// ks = 8*(l>>4)  (ISA 7.12.2, 16-bit A-matrix).
static __device__ __forceinline__ v16h frag_a(const _Float16* tile, int lane) {
  const _Float16* p = tile + (lane & 15) * 32 + ((lane >> 4) << 3);
  v8h lo = *(const v8h*)(p);
  v8h hi = *(const v8h*)(p + 16);
  return cat8(lo, hi);
}

// B fragment (32x16 f16): LDS tile row-major [k][ldn halves], n contiguous.
// Lane l holds column N=l&15; halves 0..7 = K[ks..ks+7], halves 8..15 = K[16+ks..], ks = 8*(l>>4).
static __device__ __forceinline__ v16h frag_b(const _Float16* tile, int ldn, int lane) {
  const _Float16* p = tile + (lane & 15) + ((lane >> 4) << 3) * ldn;
  v16h v;
#pragma unroll
  for (int s = 0; s < 8; ++s) { v[s] = p[s * ldn]; v[s + 8] = p[(s + 16) * ldn]; }
  return v;
}

static __device__ __forceinline__ v8f wmma16(v16h a, v16h b, v8f c) {
  return __builtin_amdgcn_wmma_f32_16x16x32_f16(false, a, false, b, (short)0, c, false, false);
}

// Async global->LDS copies (VGPR-bypass, ASYNCcnt). VDST VGPR = wave-relative
// LDS byte address (low 32 bits of the generic pointer; ISA §10.2 aperture).
// INST_OFFSET is added to BOTH the global and LDS addresses (ISA §8 async
// pseudocode), so 64 contiguous bytes need one address pair + 4 immediates.
static __device__ __forceinline__ void async_copy16(const void* gptr, void* lptr) {
  unsigned loff = (unsigned)(uintptr_t)lptr;
  asm volatile("global_load_async_to_lds_b128 %0, %1, off"
               :: "v"(loff), "v"(gptr) : "memory");
}
static __device__ __forceinline__ void async_copy64(const void* gptr, void* lptr) {
  unsigned loff = (unsigned)(uintptr_t)lptr;
  asm volatile("global_load_async_to_lds_b128 %0, %1, off\n\t"
               "global_load_async_to_lds_b128 %0, %1, off offset:16\n\t"
               "global_load_async_to_lds_b128 %0, %1, off offset:32\n\t"
               "global_load_async_to_lds_b128 %0, %1, off offset:48"
               :: "v"(loff), "v"(gptr) : "memory");
}
static __device__ __forceinline__ void wait_async_le0() {
  asm volatile("s_wait_asynccnt 0" ::: "memory");
}
static __device__ __forceinline__ void wait_async_le4() {
  asm volatile("s_wait_asynccnt 4" ::: "memory");   // oldest group of 4 landed
}

// ---------------------------------------------------------------------------
// Kernel 0: Wt[i][c] = sum_f proj_w[f] * filters[f][c][i]   (f16 output)
// ---------------------------------------------------------------------------
__global__ void __launch_bounds__(256) prep_wt(const float* __restrict__ filters,
                                               const float* __restrict__ proj_w,
                                               _Float16* __restrict__ Wt) {
  const int c = blockIdx.x;   // c_out
  const int i = threadIdx.x;  // c_in (contiguous reads)
  float s = 0.f;
#pragma unroll
  for (int f = 0; f < F_; ++f)
    s += proj_w[f] * filters[((size_t)f * COUT + c) * CIN + i];
  Wt[(size_t)i * COUT + c] = (_Float16)s;
}

// ---------------------------------------------------------------------------
// Kernel 1: X[b][n][c] = node_feats[b][n][:] . Wt[:][c] + proj_b   (f16 out)
// tile 64(node) x 64(c_out), K=256. B tile (Wt, already f16) via async->LDS.
// ---------------------------------------------------------------------------
__global__ void __launch_bounds__(256) gemm1(const float* __restrict__ nf,
                                             const _Float16* __restrict__ Wt,
                                             const float* __restrict__ proj_b,
                                             _Float16* __restrict__ X) {
  __shared__ _Float16 As[64 * 32];
  __shared__ _Float16 Bs[32 * 64];

  const int t = threadIdx.x;
  const int w = t >> 5, lane = t & 31;
  const int b  = blockIdx.z;
  const int m0 = blockIdx.y * 64;   // node block
  const int n0 = blockIdx.x * 64;   // c_out block
  const int mt  = w >> 1;
  const int ntb = (w & 1) * 2;
  const float pb = proj_b[0];

  const float*    Abase = nf + ((size_t)b * N_ + m0 + (t >> 2)) * CIN + (t & 3) * 8;
  const _Float16* Bbase = Wt + (size_t)(t >> 3) * COUT + n0 + (t & 7) * 8;
  _Float16*       Blds  = Bs + (t >> 3) * 64 + (t & 7) * 8;
  _Float16*       Alds  = As + (t >> 2) * 32 + (t & 3) * 8;

  v8f acc0 = {}; v8f acc1 = {};

  for (int it = 0; it < CIN / 32; ++it) {
    const int kk = it * 32;
    f4 a0 = *(const f4*)(Abase + kk);
    f4 a1 = *(const f4*)(Abase + kk + 4);

    __syncthreads();   // previous compute done before LDS overwrite
    async_copy16(Bbase + (size_t)kk * COUT, Blds);   // Wt tile: global->LDS direct
    {
      v8h ah;
#pragma unroll
      for (int j = 0; j < 4; ++j) { ah[j] = (_Float16)a0[j]; ah[j + 4] = (_Float16)a1[j]; }
      *(v8h*)(Alds) = ah;
    }
    wait_async_le0();
    __syncthreads();   // tiles ready

    v16h bf0 = frag_b(Bs + ntb * 16, 64, lane);
    v16h bf1 = frag_b(Bs + ntb * 16 + 16, 64, lane);
    v16h af  = frag_a(As + mt * 16 * 32, lane);
    acc0 = wmma16(af, bf0, acc0);
    acc1 = wmma16(af, bf1, acc1);
  }

  _Float16* Xb = X + (size_t)b * N_ * COUT;
  const int rbase = m0 + mt * 16 + ((lane >> 4) << 3);
  const int cb    = n0 + ntb * 16 + (lane & 15);
#pragma unroll
  for (int r = 0; r < 8; ++r) {
    Xb[(size_t)(rbase + r) * COUT + cb]      = (_Float16)(acc0[r] + pb);
    Xb[(size_t)(rbase + r) * COUT + cb + 16] = (_Float16)(acc1[r] + pb);
  }
}

// ---------------------------------------------------------------------------
// Kernel 2: out[b][m][c] = (adj[b][m][:] @ X[b][:][c]) / (rowsum(adj)+eps)
// tile 64(m) x 256(all c_out), K=2048. adj read exactly once (f32->f16 via
// VGPRs, fused row-sum); X tiles via async b128 global->LDS. Triple-buffered
// depth-2 pipeline keyed off ASYNCcnt; ONE barrier per K-step.
// ---------------------------------------------------------------------------
__global__ void __launch_bounds__(256) gemm2(const float* __restrict__ adj,
                                             const _Float16* __restrict__ X,
                                             float* __restrict__ out) {
  __shared__ _Float16 As[3][64 * 32];
  __shared__ _Float16 Bs[3][32 * 256];
  __shared__ float rowSum[64];
  __shared__ float rowScale[64];

  const int t = threadIdx.x;
  const int w = t >> 5, lane = t & 31;
  const int b  = blockIdx.y;
  const int m0 = blockIdx.x * 64;
  const int n0 = w * 32;

  const float*    Abase = adj + ((size_t)b * N_ + m0 + (t >> 2)) * (size_t)N_ + (t & 3) * 8;
  const _Float16* Bbase = X + (size_t)b * N_ * COUT + (size_t)(t >> 3) * COUT + (t & 7) * 32;
  const int aOff = (t >> 2) * 32 + (t & 3) * 8;
  const int bOff = (t >> 3) * 256 + (t & 7) * 32;

  if (t < 64) rowSum[t] = 0.f;

  v8f acc[4][2];
#pragma unroll
  for (int mt = 0; mt < 4; ++mt) { acc[mt][0] = (v8f){}; acc[mt][1] = (v8f){}; }

  float part = 0.f;
  f4 a0[2], a1[2];   // adj staging, 2 K-steps deep (slot = it & 1)

  // ---- prologue: async B for steps 0,1; A step 0 stored, A step 1 staged ----
  async_copy64(Bbase,             Bs[0] + bOff);
  async_copy64(Bbase + 32 * COUT, Bs[1] + bOff);
  a0[0] = *(const f4*)(Abase);
  a1[0] = *(const f4*)(Abase + 4);
  {
    v8h ah;
#pragma unroll
    for (int j = 0; j < 4; ++j) {
      part += a0[0][j] + a1[0][j];
      ah[j] = (_Float16)a0[0][j]; ah[j + 4] = (_Float16)a1[0][j];
    }
    *(v8h*)(As[0] + aOff) = ah;
  }
  a0[1] = *(const f4*)(Abase + 32);
  a1[1] = *(const f4*)(Abase + 36);
  wait_async_le4();   // step-0 group landed (in-order completion)
  __syncthreads();

  int cur = 0, nx1 = 1, nx2 = 2;   // rotating buffer indices
  for (int it = 0; it < N_ / 32; ++it) {
    const bool more1 = (it + 1) < (N_ / 32);
    const bool more2 = (it + 2) < (N_ / 32);

    if (more2) {
      // Bs[nx2]/A-slot were drained before barrier(it-1) -> safe to refill
      const size_t kk = (size_t)(it + 2) * 32;
      async_copy64(Bbase + kk * COUT, Bs[nx2] + bOff);
      a0[it & 1] = *(const f4*)(Abase + kk);
      a1[it & 1] = *(const f4*)(Abase + kk + 4);
    }

    // compute from current buffer (overlaps the loads above, 2 steps deep)
    v16h bf0 = frag_b(Bs[cur] + n0, 256, lane);
    v16h bf1 = frag_b(Bs[cur] + n0 + 16, 256, lane);
#pragma unroll
    for (int mt = 0; mt < 4; ++mt) {
      v16h af = frag_a(As[cur] + mt * 16 * 32, lane);
      acc[mt][0] = wmma16(af, bf0, acc[mt][0]);
      acc[mt][1] = wmma16(af, bf1, acc[mt][1]);
    }

    if (more1) {
      // publish step it+1: convert staged adj, ensure its async group landed
      const f4 x0 = a0[(it + 1) & 1], x1 = a1[(it + 1) & 1];
      v8h ah;
#pragma unroll
      for (int j = 0; j < 4; ++j) {
        part += x0[j] + x1[j];
        ah[j] = (_Float16)x0[j]; ah[j + 4] = (_Float16)x1[j];
      }
      *(v8h*)(As[nx1] + aOff) = ah;
      if (more2) wait_async_le4(); else wait_async_le0();
    }
    __syncthreads();   // single barrier per K-step

    const int tmp = cur; cur = nx1; nx1 = nx2; nx2 = tmp;
  }

  // reduce row sums (4 partials per row) via LDS float atomics
  atomicAdd(&rowSum[t >> 2], part);
  __syncthreads();
  if (t < 64) rowScale[t] = 1.f / (rowSum[t] + EPS_);
  __syncthreads();

  float* Ob = out + ((size_t)b * N_ + m0) * COUT;
  const int cb = n0 + (lane & 15);
#pragma unroll
  for (int mt = 0; mt < 4; ++mt) {
    const int rbase = mt * 16 + ((lane >> 4) << 3);
#pragma unroll
    for (int r = 0; r < 8; ++r) {
      const float s = rowScale[rbase + r];
      Ob[(size_t)(rbase + r) * COUT + cb]      = acc[mt][0][r] * s;
      Ob[(size_t)(rbase + r) * COUT + cb + 16] = acc[mt][1][r] * s;
    }
  }
}

// ---------------------------------------------------------------------------
extern "C" void kernel_launch(void* const* d_in, const int* in_sizes, int n_in,
                              void* d_out, int out_size, void* d_ws, size_t ws_size,
                              hipStream_t stream) {
  const float* nf   = (const float*)d_in[0];  // node_feats [8,2048,256]
  const float* adj  = (const float*)d_in[1];  // adj        [8,2048,2048]
  const float* filt = (const float*)d_in[2];  // filters    [8,256,256]
  const float* pw   = (const float*)d_in[3];  // proj_w     [1,8]
  const float* pb   = (const float*)d_in[4];  // proj_b     [1]
  float* out = (float*)d_out;

  _Float16* Wt = (_Float16*)d_ws;                                        // 256*256 f16
  _Float16* X  = (_Float16*)((char*)d_ws + (size_t)CIN * COUT * 2);      // 8*2048*256 f16

  prep_wt<<<COUT, CIN, 0, stream>>>(filt, pw, Wt);
  gemm1<<<dim3(COUT / 64, N_ / 64, B_), 256, 0, stream>>>(nf, Wt, pb, X);
  gemm2<<<dim3(N_ / 64, B_), 256, 0, stream>>>(adj, X, out);
}
